// VNGNN_53721450938782
// MI455X (gfx1250) — matrix operand order repaired and checked by reading, not compiled, not run on hardware
//
#include <hip/hip_runtime.h>
#include <hip/hip_bf16.h>
#include <math.h>

// Problem sizes (fixed by the reference)
constexpr int N = 262144;
constexpr int C = 1024;
constexpr int D = 128;

// Padded LDS strides (in halfs): +8 halfs (16B) => rows shift 4 LDS banks,
// keeps 16-byte alignment for ds_load_b128.
constexpr int SA = D + 8;   // h tile stride
constexpr int SB = D + 8;   // transposed weight tile stride

typedef __attribute__((ext_vector_type(16))) __bf16 v16bf;
typedef __attribute__((ext_vector_type(8)))  __bf16 v8bf;
typedef __attribute__((ext_vector_type(8)))  float  v8f;

// Hardware tanh (V_TANH_F32, TRANS32 pipe) — confirmed emitted on gfx1250.
__device__ __forceinline__ float fast_tanh(float x) {
#if __has_builtin(__builtin_amdgcn_tanhf)
  return __builtin_amdgcn_tanhf(x);
#elif __has_builtin(__builtin_amdgcn_tanh_f32)
  return __builtin_amdgcn_tanh_f32(x);
#else
  return tanhf(x);
#endif
}

__device__ __forceinline__ float fast_sigmoid(float x) {
  return 1.0f / (1.0f + __expf(-x));
}

// A-fragment (16x32 bf16, wave32 ISA layout): lane holds row m = lane&15,
// k = (v<4?0:16) + 8*hi + e  -> two contiguous 8-half (16B) runs.
__device__ __forceinline__ v16bf load_a_frag(const __bf16* __restrict__ arow,
                                             int kc, int hi) {
  const __bf16* p = arow + kc * 32 + hi * 8;
  v8bf lo = *(const v8bf*)p;          // k = kc*32 + 8*hi + [0..7]
  v8bf up = *(const v8bf*)(p + 16);   // k = kc*32 + 16 + 8*hi + [0..7]
  return __builtin_shufflevector(lo, up, 0, 1, 2, 3, 4, 5, 6, 7,
                                 8, 9, 10, 11, 12, 13, 14, 15);
}

// B-fragment (32x16 bf16): lane holds col n = lane&15,
// k = (v<4?0:16) + 8*hi + e -> two 8-half runs in the TRANSPOSED weight tile (sBt[n][k]).
__device__ __forceinline__ v16bf load_b_frag(const __bf16* __restrict__ brow,
                                             int kc, int hi) {
  const __bf16* p = brow + kc * 32 + hi * 8;
  v8bf lo = *(const v8bf*)p;
  v8bf up = *(const v8bf*)(p + 16);
  return __builtin_shufflevector(lo, up, 0, 1, 2, 3, 4, 5, 6, 7,
                                 8, 9, 10, 11, 12, 13, 14, 15);
}

// ---------------------------------------------------------------------------
__global__ void __launch_bounds__(256)
init_kernel(float* __restrict__ ctx_u, float* __restrict__ denom,
            unsigned* __restrict__ m_enc) {
  int i = blockIdx.x * blockDim.x + threadIdx.x;
  if (i < C * D) ctx_u[i] = 0.0f;
  if (i < C) { denom[i] = 0.0f; m_enc[i] = 0u; }  // enc 0 acts as -inf sentinel
}

// qg[c][d] = sum_k g[c][k] * Wq[k][d] + attn_bias[d]
__global__ void __launch_bounds__(128)
qg_kernel(const float* __restrict__ g, const float* __restrict__ Wq,
          const float* __restrict__ attn_bias, float* __restrict__ qg) {
  __shared__ float gr[D];
  const int c = blockIdx.x, d = threadIdx.x;
  gr[d] = g[c * D + d];
  __syncthreads();
  float acc = attn_bias[d];
  #pragma unroll 8
  for (int k = 0; k < D; ++k) acc += gr[k] * Wq[k * D + d];
  qg[c * D + d] = acc;
}

// ---------------------------------------------------------------------------
// WMMA score pass: 64 rows/block, 4 waves. score_i = tanh(h_i@Wk + qg[ca_i]).w + b0
// ---------------------------------------------------------------------------
__global__ void __launch_bounds__(128)
score_kernel(const float* __restrict__ h, const float* __restrict__ Wk,
             const float* __restrict__ qg, const int* __restrict__ ca,
             const float* __restrict__ w_score, const float* __restrict__ b_score,
             float* __restrict__ score, unsigned* __restrict__ m_enc) {
  __shared__ __bf16 sA[64 * SA];   // h tile, row-major, padded
  __shared__ __bf16 sBt[D * SB];   // Wk TRANSPOSED: sBt[n][k], padded
  __shared__ float  sScore[64];

  const int tid  = threadIdx.x;
  const int lane = tid & 31;
  const int wave = tid >> 5;            // 0..3, 16 rows each
  const int rowBase = blockIdx.x * 64;

  for (int i = tid; i < D * D; i += 128) {      // coalesced global read
    const int k = i >> 7, n = i & 127;
    sBt[n * SB + k] = (__bf16)Wk[i];            // transposed, padded write
  }
  for (int i = tid; i < 64 * D; i += 128) {
    const int r = i >> 7, k = i & 127;
    sA[r * SA + k] = (__bf16)h[(size_t)rowBase * D + i];
  }
  if (tid < 64) sScore[tid] = 0.0f;
  __syncthreads();

  const int m  = lane & 15;
  const int hi = lane >> 4;
  const int warpRow = wave * 16;
  const __bf16* arow = &sA[(warpRow + m) * SA];

  v16bf aF[4];
  #pragma unroll
  for (int kc = 0; kc < 4; ++kc) aF[kc] = load_a_frag(arow, kc, hi);

  int crow[8];
  #pragma unroll
  for (int r = 0; r < 8; ++r) crow[r] = ca[rowBase + warpRow + r + 8 * hi];

  #pragma unroll 2
  for (int ct = 0; ct < 8; ++ct) {      // 8 column tiles of 16
    const __bf16* brow = &sBt[(ct * 16 + m) * SB];
    v8f acc = {0.f, 0.f, 0.f, 0.f, 0.f, 0.f, 0.f, 0.f};
    #pragma unroll
    for (int kc = 0; kc < 4; ++kc) {
      v16bf bF = load_b_frag(brow, kc, hi);
      acc = __builtin_amdgcn_wmma_f32_16x16x32_bf16(
          false, aF[kc], false, bF, (short)0, acc, false, false);
    }
    const int n = ct * 16 + m;
    // Batch the 8 L2-resident gathers so they overlap (one wait, not eight).
    float qv[8];
    #pragma unroll
    for (int r = 0; r < 8; ++r) qv[r] = qg[(crow[r] << 7) + n];
    const float wsn = w_score[n];
    #pragma unroll
    for (int r = 0; r < 8; ++r) {
      const int row = warpRow + r + 8 * hi;
      const float t = fast_tanh(acc[r] + qv[r]);
      atomicAdd(&sScore[row], t * wsn);
    }
  }
  __syncthreads();

  if (tid < 64) {
    const float s = sScore[tid] + b_score[0];
    const int i = rowBase + tid;
    score[i] = s;
    const unsigned bits = __float_as_uint(s);
    const unsigned enc = (s >= 0.0f) ? (bits | 0x80000000u) : ~bits;
    atomicMax(&m_enc[ca[i]], enc);
  }
}

// ---------------------------------------------------------------------------
// e = exp(score - max[ca]); denom[ca] += e; ctx_u[ca] += e*h_i  (L2-resident)
// ---------------------------------------------------------------------------
__global__ void __launch_bounds__(256)
exp_ctx_kernel(const float* __restrict__ h, const float* __restrict__ score,
               const unsigned* __restrict__ m_enc, const int* __restrict__ ca,
               float* __restrict__ denom, float* __restrict__ ctx_u) {
  const int node = blockIdx.x * 2 + (threadIdx.x >> 7);
  const int d    = threadIdx.x & 127;
  const int c    = ca[node];
  const unsigned enc = m_enc[c];
  const float mval = (enc & 0x80000000u) ? __uint_as_float(enc & 0x7FFFFFFFu)
                                         : __uint_as_float(~enc);
  const float e = __expf(score[node] - mval);
  if (d == 0) atomicAdd(&denom[c], e);
  atomicAdd(&ctx_u[(size_t)c * D + d], e * h[(size_t)node * D + d]);
}

// ---------------------------------------------------------------------------
// Fused cluster chain (one block per cluster, thread = output dim).
// ---------------------------------------------------------------------------
__global__ void __launch_bounds__(128)
cluster_kernel(const float* __restrict__ g, const float* __restrict__ denom,
               const float* __restrict__ ctx_u,
               const float* __restrict__ W_gn, const float* __restrict__ b_gn,
               const float* __restrict__ W_upd, const float* __restrict__ b_upd,
               const float* __restrict__ H_vn, const float* __restrict__ bH_vn,
               const float* __restrict__ G_vn, const float* __restrict__ bG_vn,
               const float* __restrict__ F_sup, const float* __restrict__ bF_sup,
               const float* __restrict__ G_gn, const float* __restrict__ bG_gn,
               float* __restrict__ g_trans, float* __restrict__ A_gate) {
  __shared__ float ctx[D], ht[D], gr[D], gu[D], gn2[D], gt[D];
  const int c = blockIdx.x, d = threadIdx.x;

  const float dn = denom[c];
  ctx[d] = ctx_u[c * D + d] / dn;
  gr[d]  = g[c * D + d];
  __syncthreads();

  float a = b_gn[d], b = b_upd[d];
  #pragma unroll 8
  for (int k = 0; k < D; ++k) {
    a += ctx[k] * W_gn[k * D + d];
    b += gr[k]  * W_upd[k * D + d];
  }
  ht[d] = fast_tanh(a);
  gu[d] = b;
  __syncthreads();

  float zacc = bH_vn[d] + bG_vn[d];
  #pragma unroll 8
  for (int k = 0; k < D; ++k)
    zacc += gu[k] * H_vn[k * D + d] + ht[k] * G_vn[k * D + d];
  const float z = fast_sigmoid(zacc);
  gn2[d] = (1.0f - z) * gu[d] + z * ht[d];
  __syncthreads();

  float f = bF_sup[d];
  #pragma unroll 8
  for (int k = 0; k < D; ++k) f += gn2[k] * F_sup[k * D + d];
  gt[d] = fast_tanh(f);
  __syncthreads();

  float ag = bG_gn[d];
  #pragma unroll 8
  for (int k = 0; k < D; ++k) ag += gt[k] * G_gn[k * D + d];
  g_trans[c * D + d] = gt[d];
  A_gate[c * D + d]  = ag;
}

// ---------------------------------------------------------------------------
// WMMA final pass: z = sigmoid(h@H_gn + bH_gn + A_gate[ca]);
// h_new = (1-z)*h + z*g_trans[ca]
// ---------------------------------------------------------------------------
__global__ void __launch_bounds__(128)
final_kernel(const float* __restrict__ h, const float* __restrict__ Hgn,
             const float* __restrict__ bHgn, const float* __restrict__ A_gate,
             const float* __restrict__ g_trans, const int* __restrict__ ca,
             float* __restrict__ out) {
  __shared__ __bf16 sA[64 * SA];
  __shared__ __bf16 sBt[D * SB];

  const int tid  = threadIdx.x;
  const int lane = tid & 31;
  const int wave = tid >> 5;
  const int rowBase = blockIdx.x * 64;

  for (int i = tid; i < D * D; i += 128) {
    const int k = i >> 7, n = i & 127;
    sBt[n * SB + k] = (__bf16)Hgn[i];
  }
  for (int i = tid; i < 64 * D; i += 128) {
    const int r = i >> 7, k = i & 127;
    sA[r * SA + k] = (__bf16)h[(size_t)rowBase * D + i];
  }
  __syncthreads();

  const int m  = lane & 15;
  const int hi = lane >> 4;
  const int warpRow = wave * 16;
  const __bf16* arow = &sA[(warpRow + m) * SA];

  v16bf aF[4];
  #pragma unroll
  for (int kc = 0; kc < 4; ++kc) aF[kc] = load_a_frag(arow, kc, hi);

  int crow[8];
  #pragma unroll
  for (int r = 0; r < 8; ++r) crow[r] = ca[rowBase + warpRow + r + 8 * hi];

  #pragma unroll 2
  for (int ct = 0; ct < 8; ++ct) {
    const __bf16* brow = &sBt[(ct * 16 + m) * SB];
    v8f acc = {0.f, 0.f, 0.f, 0.f, 0.f, 0.f, 0.f, 0.f};
    #pragma unroll
    for (int kc = 0; kc < 4; ++kc) {
      v16bf bF = load_b_frag(brow, kc, hi);
      acc = __builtin_amdgcn_wmma_f32_16x16x32_bf16(
          false, aF[kc], false, bF, (short)0, acc, false, false);
    }
    const int n = ct * 16 + m;
    // Batch all per-tile gathers (A_gate, g_trans: L2-resident; h: L2 hit),
    // then run the TRANS/blend chain with everything in flight.
    float av[8], gv[8], hv[8];
    #pragma unroll
    for (int r = 0; r < 8; ++r) av[r] = A_gate[(crow[r] << 7) + n];
    #pragma unroll
    for (int r = 0; r < 8; ++r) gv[r] = g_trans[(crow[r] << 7) + n];
    #pragma unroll
    for (int r = 0; r < 8; ++r) {
      const int row = warpRow + r + 8 * hi;
      hv[r] = h[(size_t)(rowBase + row) * D + n];
    }
    const float bn = bHgn[n];
    #pragma unroll
    for (int r = 0; r < 8; ++r) {
      const int row = warpRow + r + 8 * hi;
      const float z = fast_sigmoid(acc[r] + bn + av[r]);
      out[(size_t)(rowBase + row) * D + n] = (1.0f - z) * hv[r] + z * gv[r];
    }
  }
}

// ---------------------------------------------------------------------------
extern "C" void kernel_launch(void* const* d_in, const int* in_sizes, int n_in,
                              void* d_out, int out_size, void* d_ws, size_t ws_size,
                              hipStream_t stream) {
  const float* h         = (const float*)d_in[0];
  const float* g         = (const float*)d_in[1];
  const int*   ca        = (const int*)  d_in[2];
  const float* Wq        = (const float*)d_in[3];
  const float* Wk        = (const float*)d_in[4];
  const float* attn_bias = (const float*)d_in[5];
  const float* w_score   = (const float*)d_in[6];
  const float* b_score   = (const float*)d_in[7];
  const float* W_gn      = (const float*)d_in[8];
  const float* b_gn      = (const float*)d_in[9];
  const float* W_upd     = (const float*)d_in[10];
  const float* b_upd     = (const float*)d_in[11];
  const float* H_vn      = (const float*)d_in[12];
  const float* bH_vn     = (const float*)d_in[13];
  const float* G_vn      = (const float*)d_in[14];
  const float* bG_vn     = (const float*)d_in[15];
  const float* F_sup     = (const float*)d_in[16];
  const float* bF_sup    = (const float*)d_in[17];
  const float* H_gn      = (const float*)d_in[18];
  const float* bH_gn     = (const float*)d_in[19];
  const float* G_gn      = (const float*)d_in[20];
  const float* bG_gn     = (const float*)d_in[21];
  float* out = (float*)d_out;

  char* p = (char*)d_ws;
  float*    qg      = (float*)p;    p += (size_t)C * D * sizeof(float);
  float*    score   = (float*)p;    p += (size_t)N * sizeof(float);
  unsigned* m_enc   = (unsigned*)p; p += (size_t)C * sizeof(unsigned);
  float*    denom   = (float*)p;    p += (size_t)C * sizeof(float);
  float*    ctx_u   = (float*)p;    p += (size_t)C * D * sizeof(float);
  float*    g_trans = (float*)p;    p += (size_t)C * D * sizeof(float);
  float*    A_gate  = (float*)p;    p += (size_t)C * D * sizeof(float);

  init_kernel<<<(C * D + 255) / 256, 256, 0, stream>>>(ctx_u, denom, m_enc);
  qg_kernel<<<C, D, 0, stream>>>(g, Wq, attn_bias, qg);
  score_kernel<<<N / 64, 128, 0, stream>>>(h, Wk, qg, ca, w_score, b_score,
                                           score, m_enc);
  exp_ctx_kernel<<<N / 2, 256, 0, stream>>>(h, score, m_enc, ca, denom, ctx_u);
  cluster_kernel<<<C, D, 0, stream>>>(g, denom, ctx_u, W_gn, b_gn, W_upd, b_upd,
                                      H_vn, bH_vn, G_vn, bG_vn, F_sup, bF_sup,
                                      G_gn, bG_gn, g_trans, A_gate);
  final_kernel<<<N / 64, 128, 0, stream>>>(h, H_gn, bH_gn, A_gate, g_trans, ca, out);
}
